// NEGAPooling_25615184953984
// MI455X (gfx1250) — compile-verified, matrix operand order checked by reading, not run on hardware
//
#include <hip/hip_runtime.h>

// ---------------------------------------------------------------------------
// NEGAPooling for MI455X (gfx1250, wave32, WMMA bf16)
// B=16, T=2048, D=256, QK=64
// ---------------------------------------------------------------------------

typedef __bf16 bf16;
typedef __attribute__((ext_vector_type(16))) __bf16       v16bf;
typedef __attribute__((ext_vector_type(8)))  float        v8f;
typedef __attribute__((ext_vector_type(4)))  unsigned int u32x4;

#define BB 16
#define TT 2048
#define DD 256
#define QQ 64

struct BF16Tile { u32x4 lo, hi; };

// Load a 16x32 bf16 WMMA operand tile.
// A-matrix layout (ISA 7.12.2, 16-bit A 16x32): lanes 0-15 hold row M=lane,
// K chunks {0..7, 16..23}; lanes 16-31 hold row M=lane-16, K {8..15, 24..31}.
// We use the mirrored pattern for B operands (lane = column N, same K chunks),
// which requires B memory to be [N, K] row-major (K contiguous per column).
__device__ __forceinline__ v16bf load_tile(const bf16* base, int stride) {
  const int lane = threadIdx.x & 31;
  const bf16* p = base + (size_t)(lane & 15) * stride + ((lane >> 4) << 3);
  BF16Tile t;
  t.lo = *(const u32x4*)(p);
  t.hi = *(const u32x4*)(p + 16);
  return __builtin_bit_cast(v16bf, t);
}

__device__ __forceinline__ float rmax16(float v) {
  v = fmaxf(v, __shfl_xor(v, 1));
  v = fmaxf(v, __shfl_xor(v, 2));
  v = fmaxf(v, __shfl_xor(v, 4));
  v = fmaxf(v, __shfl_xor(v, 8));
  return v;
}
__device__ __forceinline__ float rsum16(float v) {
  v += __shfl_xor(v, 1);
  v += __shfl_xor(v, 2);
  v += __shfl_xor(v, 4);
  v += __shfl_xor(v, 8);
  return v;
}

// ------------------------------- prep kernels ------------------------------

__global__ void zero_kernel(float* __restrict__ p, int n) {
  int i = blockIdx.x * blockDim.x + threadIdx.x;
  if (i < n) p[i] = 0.0f;
}

__global__ void cvt_kernel(const float* __restrict__ src, bf16* __restrict__ dst, int n) {
  int i = blockIdx.x * blockDim.x + threadIdx.x;
  if (i < n) dst[i] = (bf16)src[i];
}

__global__ void maskbias_kernel(const int* __restrict__ mask, float* __restrict__ mb, int n) {
  int i = blockIdx.x * blockDim.x + threadIdx.x;
  if (i < n) mb[i] = (mask[i] == 0) ? -1e9f : 0.0f;
}

// --------------------------- projection kernel -----------------------------
// Computes [B*T, 384] = xb[B*T,256] @ WBᵀ (WB = [Wq;Wk;Wv] row-major [384,256])
// cols 0..63 -> q (bf16, [B*T,64])
// cols 64..127 -> k scaled by (1/sqrt(QK))*(1+clip(sl)^2)  (bf16, [B*T,64])
// cols 128..383 -> v, stored transposed as vt[b][d][t] (bf16)
__global__ __launch_bounds__(128) void proj_kernel(
    const bf16* __restrict__ xb, const bf16* __restrict__ wb,
    const float* __restrict__ bq, const float* __restrict__ bk,
    const float* __restrict__ bv, const float* __restrict__ sl,
    bf16* __restrict__ qb, bf16* __restrict__ kb, bf16* __restrict__ vtb)
{
  const int lane = threadIdx.x & 31;
  const int wave = threadIdx.x >> 5;
  const int tile = blockIdx.x;                 // 0..(B*T/16 - 1): 16-row tile
  const int nt   = blockIdx.y * 4 + wave;      // 0..23: 16-col tile
  const int e0   = nt * 16;
  const int n    = lane & 15, hi = lane >> 4;
  const int col  = e0 + n;

  float bias = (col < QQ) ? bq[col] : (col < 2*QQ ? bk[col - QQ] : bv[col - 2*QQ]);
  v8f c;
  for (int r = 0; r < 8; ++r) c[r] = bias;

  const bf16* arow = xb + (size_t)tile * 16 * DD;
  const bf16* brow = wb + (size_t)e0 * DD;
  for (int kk = 0; kk < DD; kk += 32) {
    v16bf a = load_tile(arow + kk, DD);
    v16bf w = load_tile(brow + kk, DD);
    c = __builtin_amdgcn_wmma_f32_16x16x32_bf16(false, a, false, w, (short)0, c, false, false);
  }

  // Epilogue. C/D layout: lane holds column n; VGPR r holds row (r + 8*hi).
  if (nt < 8) {
    const bool isk = (nt >= 4);
    for (int r = 0; r < 8; ++r) {
      int gr = tile * 16 + r + 8 * hi;         // global row in [0, B*T)
      float v = c[r];
      if (isk) {
        float s = sl[gr];
        s = fminf(fmaxf(s, 0.0f), 1.0f);
        v *= 0.125f * (1.0f + s * s);          // 1/sqrt(64) * (1 + sl^2)
        kb[(size_t)gr * QQ + (col - QQ)] = (bf16)v;
      } else {
        qb[(size_t)gr * QQ + col] = (bf16)v;
      }
    }
  } else {
    int d  = col - 2 * QQ;
    int b  = tile >> 7;                        // tile / (T/16)
    int t0 = ((tile & 127) << 4) + 8 * hi;     // 8 consecutive t values
    __attribute__((aligned(16))) bf16 pk[8];
    for (int r = 0; r < 8; ++r) pk[r] = (bf16)c[r];
    *(u32x4*)(vtb + ((size_t)(b * DD + d)) * TT + t0) = *(const u32x4*)pk;
  }
}

// ---------------------------- attention kernel -----------------------------
// grid (T/16, B), block 128 = 4 waves. Wave w owns 16 queries x 64 out cols.
__global__ __launch_bounds__(128) void attn_kernel(
    const bf16* __restrict__ qb, const bf16* __restrict__ kb,
    const bf16* __restrict__ vtb, const float* __restrict__ mbias,
    float* __restrict__ out)
{
  __shared__ __align__(16) bf16 plds[4][16 * 40];   // 80B row stride per wave
  const int lane = threadIdx.x & 31;
  const int wave = threadIdx.x >> 5;
  const int n = lane & 15, hi = lane >> 4;
  const int qt = blockIdx.x;
  const int b  = blockIdx.y;
  const int d0 = wave * 64;

  const size_t qrow = (size_t)(b * TT + qt * 16) * QQ;
  v16bf qa0 = load_tile(qb + qrow + 0,  QQ);
  v16bf qa1 = load_tile(qb + qrow + 32, QQ);

  v8f O0 = {}, O1 = {}, O2 = {}, O3 = {};
  float m[8], l[8];
  for (int r = 0; r < 8; ++r) { m[r] = -1e30f; l[r] = 0.0f; }

  bf16* my = &plds[wave][0];

  for (int j0 = 0; j0 < TT; j0 += 32) {
    const size_t krow0 = (size_t)(b * TT + j0) * QQ;
    const size_t krow1 = (size_t)(b * TT + j0 + 16) * QQ;
    v16bf k0a = load_tile(kb + krow0 + 0,  QQ);
    v16bf k0b = load_tile(kb + krow0 + 32, QQ);
    v16bf k1a = load_tile(kb + krow1 + 0,  QQ);
    v16bf k1b = load_tile(kb + krow1 + 32, QQ);

    v8f s0 = {}, s1 = {};
    s0 = __builtin_amdgcn_wmma_f32_16x16x32_bf16(false, qa0, false, k0a, (short)0, s0, false, false);
    s0 = __builtin_amdgcn_wmma_f32_16x16x32_bf16(false, qa1, false, k0b, (short)0, s0, false, false);
    s1 = __builtin_amdgcn_wmma_f32_16x16x32_bf16(false, qa0, false, k1a, (short)0, s1, false, false);
    s1 = __builtin_amdgcn_wmma_f32_16x16x32_bf16(false, qa1, false, k1b, (short)0, s1, false, false);

    float mb0 = mbias[b * TT + j0 + n];
    float mb1 = mbias[b * TT + j0 + 16 + n];

    float p0[8], p1[8];
    for (int r = 0; r < 8; ++r) { s0[r] += mb0; s1[r] += mb1; }
    for (int r = 0; r < 8; ++r) {
      float rm = rmax16(fmaxf(s0[r], s1[r]));
      float mn = fmaxf(m[r], rm);
      float corr = __expf(m[r] - mn);
      p0[r] = __expf(s0[r] - mn);
      p1[r] = __expf(s1[r] - mn);
      float rs = rsum16(p0[r] + p1[r]);
      l[r] = l[r] * corr + rs;
      m[r] = mn;
      O0[r] *= corr; O1[r] *= corr; O2[r] *= corr; O3[r] *= corr;
    }

    // Relayout P (C-layout) -> A-layout via this wave's private LDS slice.
    for (int r = 0; r < 8; ++r) {
      int row = r + 8 * hi;
      my[row * 40 + n]      = (bf16)p0[r];
      my[row * 40 + 16 + n] = (bf16)p1[r];
    }
    asm volatile("s_wait_dscnt 0x0" ::: "memory");
    v16bf pa = load_tile(my, 40);

    const size_t vrow = (size_t)(b * DD + d0) * TT + j0;
    v16bf v0 = load_tile(vtb + vrow + (size_t)0  * TT, TT);
    v16bf v1 = load_tile(vtb + vrow + (size_t)16 * TT, TT);
    v16bf v2 = load_tile(vtb + vrow + (size_t)32 * TT, TT);
    v16bf v3 = load_tile(vtb + vrow + (size_t)48 * TT, TT);

    O0 = __builtin_amdgcn_wmma_f32_16x16x32_bf16(false, pa, false, v0, (short)0, O0, false, false);
    O1 = __builtin_amdgcn_wmma_f32_16x16x32_bf16(false, pa, false, v1, (short)0, O1, false, false);
    O2 = __builtin_amdgcn_wmma_f32_16x16x32_bf16(false, pa, false, v2, (short)0, O2, false, false);
    O3 = __builtin_amdgcn_wmma_f32_16x16x32_bf16(false, pa, false, v3, (short)0, O3, false, false);
  }

  // Finalize: divide by l, sum over 16 query rows, scale by 1/T, accumulate.
  float inv[8];
  for (int r = 0; r < 8; ++r) inv[r] = 1.0f / l[r];
  const float invT = 1.0f / (float)TT;

  auto emit = [&](v8f o, int ct) {
    float s = 0.0f;
    for (int r = 0; r < 8; ++r) s += o[r] * inv[r];
    s += __shfl_xor(s, 16);                    // combine rows 0-7 with 8-15
    if (lane < 16)
      atomicAdd(&out[b * DD + d0 + ct * 16 + lane], s * invT);
  };
  emit(O0, 0); emit(O1, 1); emit(O2, 2); emit(O3, 3);
}

// ------------------------------- launcher ----------------------------------

extern "C" void kernel_launch(void* const* d_in, const int* in_sizes, int n_in,
                              void* d_out, int out_size, void* d_ws, size_t ws_size,
                              hipStream_t stream) {
  const float* x    = (const float*)d_in[0];
  const float* sl   = (const float*)d_in[1];
  const int*   mask = (const int*)  d_in[2];
  const float* Wq   = (const float*)d_in[3];
  const float* bq   = (const float*)d_in[4];
  const float* Wk   = (const float*)d_in[5];
  const float* bk   = (const float*)d_in[6];
  const float* Wv   = (const float*)d_in[7];
  const float* bv   = (const float*)d_in[8];
  float* out = (float*)d_out;

  char* ws = (char*)d_ws;
  const size_t N_X  = (size_t)BB * TT * DD;        // 8,388,608
  const size_t OFF_XB = 0;
  const size_t OFF_WB = OFF_XB + N_X * 2;          // xb: bf16
  const size_t OFF_QB = OFF_WB + (size_t)384 * DD * 2;
  const size_t OFF_KB = OFF_QB + (size_t)BB * TT * QQ * 2;
  const size_t OFF_VT = OFF_KB + (size_t)BB * TT * QQ * 2;
  const size_t OFF_MB = OFF_VT + (size_t)BB * DD * TT * 2;

  bf16*  xb  = (bf16*)(ws + OFF_XB);
  bf16*  wb  = (bf16*)(ws + OFF_WB);
  bf16*  qbp = (bf16*)(ws + OFF_QB);
  bf16*  kbp = (bf16*)(ws + OFF_KB);
  bf16*  vtb = (bf16*)(ws + OFF_VT);
  float* mb  = (float*)(ws + OFF_MB);

  // 1) zero output (harness poisons it)
  zero_kernel<<<(out_size + 255) / 256, 256, 0, stream>>>(out, out_size);

  // 2) f32 -> bf16 conversions
  cvt_kernel<<<(int)((N_X + 255) / 256), 256, 0, stream>>>(x, xb, (int)N_X);
  cvt_kernel<<<(QQ * DD + 255) / 256, 256, 0, stream>>>(Wq, wb, QQ * DD);
  cvt_kernel<<<(QQ * DD + 255) / 256, 256, 0, stream>>>(Wk, wb + QQ * DD, QQ * DD);
  cvt_kernel<<<(DD * DD + 255) / 256, 256, 0, stream>>>(Wv, wb + 2 * QQ * DD, DD * DD);

  // 3) mask bias
  maskbias_kernel<<<(BB * TT + 255) / 256, 256, 0, stream>>>(mask, mb, BB * TT);

  // 4) QKV projection (WMMA bf16)
  proj_kernel<<<dim3(BB * TT / 16, 6), 128, 0, stream>>>(
      xb, wb, bq, bk, bv, sl, qbp, kbp, vtb);

  // 5) flash attention + query-mean (WMMA bf16)
  attn_kernel<<<dim3(TT / 16, BB), 128, 0, stream>>>(qbp, kbp, vtb, mb, out);
}
